// MultiHeadAttention_1116691497080
// MI455X (gfx1250) — compile-verified, matrix operand order checked by reading, not run on hardware
//
#include <hip/hip_runtime.h>
#include <hip/hip_bf16.h>

#define B_ 8
#define S_ 2048
#define D_ 1024
#define H_ 8
#define CH_ 128
#define GROUPS_ (B_ * H_)   /* 64 */
#define MROWS_ (B_ * S_)    /* 16384 */
#define SCALE_ 0.03125f     /* 1024^-0.5 (key_dim == D in reference) */
#define LN_EPS_ 1e-5f

// ---- CDNA5 async global->LDS copies (ASYNCcnt path), with safe fallback ----
#if defined(__has_builtin)
#if __has_builtin(__builtin_amdgcn_global_load_async_to_lds_b128) && \
    __has_builtin(__builtin_amdgcn_s_wait_asynccnt)
#define HAVE_ASYNC_LDS 1
#endif
#endif
#ifndef HAVE_ASYNC_LDS
#define HAVE_ASYNC_LDS 0
#endif

typedef __attribute__((ext_vector_type(16))) __bf16 v16bf;
typedef __attribute__((ext_vector_type(8)))  __bf16 bf16x8;
typedef __attribute__((ext_vector_type(4)))  __bf16 bf16x4;
typedef __attribute__((ext_vector_type(8)))  float  v8f;
typedef __attribute__((ext_vector_type(4)))  float  f32x4;

// Matches the builtin's parameter pointee type exactly:
//   '__attribute__((__vector_size__(4 * sizeof(int)))) int'
typedef int i32x4v __attribute__((vector_size(4 * sizeof(int))));

__device__ __forceinline__ void cp16_g2l(void* lds, const void* g) {
#if HAVE_ASYNC_LDS
  __builtin_amdgcn_global_load_async_to_lds_b128(
      (__attribute__((address_space(1))) i32x4v*)(g),
      (__attribute__((address_space(3))) i32x4v*)(lds), 0, 0);
#else
  *(bf16x8*)lds = *(const bf16x8*)g;
#endif
}

__device__ __forceinline__ void cp_g2l_wait() {
#if HAVE_ASYNC_LDS
  __builtin_amdgcn_s_wait_asynccnt(0);
#endif
}

__device__ __forceinline__ __bf16 cvt_bf16(float f) {
  unsigned u = __builtin_bit_cast(unsigned, f);
  u += 0x7fffu + ((u >> 16) & 1u);            // round-to-nearest-even
  unsigned short h = (unsigned short)(u >> 16);
  return __builtin_bit_cast(__bf16, h);
}

__device__ __forceinline__ bf16x4 cvt_bf16x4(f32x4 v) {
  bf16x4 r;
  r.x = cvt_bf16(v.x); r.y = cvt_bf16(v.y);
  r.z = cvt_bf16(v.z); r.w = cvt_bf16(v.w);
  return r;
}

__device__ __forceinline__ v8f v8f_zero() {
  v8f z;
#pragma unroll
  for (int i = 0; i < 8; ++i) z[i] = 0.f;
  return z;
}

// One 16x32 bf16 WMMA operand fragment from a row-major LDS tile.
// CDNA5 wave32: A(16x32) and B(32x16) layouts are mirror images, both are
// "lane holds one row, K-contiguous, split per 16-lane half":
//   lane<16 : K {0..7} (VGPR0-3) and K {16..23} (VGPR4-7)
//   lane>=16: K {8..15}          and K {24..31}
// p0 -> element (row0, k0); ld = row stride in elements (multiple of 8).
__device__ __forceinline__ v16bf frag_ld(const __bf16* p0, int ld) {
  const int lane = (int)(threadIdx.x & 31u);
  const __bf16* p = p0 + (lane & 15) * ld + (lane >> 4) * 8;
  bf16x8 lo = *(const bf16x8*)(p);        // two ds_load_b128
  bf16x8 hi = *(const bf16x8*)(p + 16);
  return __builtin_shufflevector(lo, hi, 0,1,2,3,4,5,6,7,8,9,10,11,12,13,14,15);
}

__device__ __forceinline__ v8f wmma32(v16bf a, v16bf b, v8f c) {
  return __builtin_amdgcn_wmma_f32_16x16x32_bf16(
      false, a, false, b, (short)0, c, false, false);
}

// ---------------------------------------------------------------------------
// Kernel 1: Q/K/V projections. y = x @ W.T  (W is [out,in] row-major).
// blockIdx.z: 0 -> Q(query,Wq), 1 -> K(keys,Wk), 2 -> V(keys,Wv, stored
// transposed per attention group: Vt[g][c(128)][s(2048)]).
// Tiles: 128(M) x 128(N) x 32(K); 8 waves, each 32x64 (2x4 WMMA tiles).
// ---------------------------------------------------------------------------
__global__ __launch_bounds__(256) void qkv_kernel(
    const float* __restrict__ query, const float* __restrict__ keys,
    const float* __restrict__ Wq, const float* __restrict__ Wk,
    const float* __restrict__ Wv,
    __bf16* __restrict__ Qo, __bf16* __restrict__ Ko, __bf16* __restrict__ Vt)
{
  __shared__ __bf16 As[128 * 40];   // 128 rows x 32 cols, pad to 40
  __shared__ __bf16 Bs[128 * 40];

  const int t = (int)threadIdx.x;
  const int lane = t & 31, wave = t >> 5;
  const int wm = wave & 3, wn = wave >> 2;      // 4x2 wave grid
  const int m0 = (int)blockIdx.x * 128;
  const int n0 = (int)blockIdx.y * 128;
  const int z = (int)blockIdx.z;
  const float* X = (z == 0) ? query : keys;
  const float* W = (z == 0) ? Wq : (z == 1) ? Wk : Wv;

  v8f acc[2][4];
#pragma unroll
  for (int i = 0; i < 2; ++i)
#pragma unroll
    for (int j = 0; j < 4; ++j) acc[i][j] = v8f_zero();

  for (int k0 = 0; k0 < D_; k0 += 32) {
    // Batch all 8 staging loads first (overlap latency), then convert.
    f32x4 av[4], wv[4];
#pragma unroll
    for (int i = 0; i < 4; ++i) {
      int e = t + i * 256;                 // 1024 float4 slots = 128 x 8
      int r = e >> 3, c4 = e & 7;
      av[i] = *(const f32x4*)(X + (size_t)(m0 + r) * D_ + k0 + c4 * 4);
      wv[i] = *(const f32x4*)(W + (size_t)(n0 + r) * D_ + k0 + c4 * 4);
    }
#pragma unroll
    for (int i = 0; i < 4; ++i) {
      int e = t + i * 256;
      int r = e >> 3, c4 = e & 7;
      *(bf16x4*)(As + r * 40 + c4 * 4) = cvt_bf16x4(av[i]);
      *(bf16x4*)(Bs + r * 40 + c4 * 4) = cvt_bf16x4(wv[i]);
    }
    if (k0 + 32 < D_) {                    // global_prefetch_b8 next tiles
      __builtin_prefetch(X + (size_t)(m0 + (t >> 1)) * D_ + k0 + 32, 0, 1);
      __builtin_prefetch(W + (size_t)(n0 + (t >> 1)) * D_ + k0 + 32, 0, 1);
    }
    __syncthreads();

    v16bf aF[2], bF[4];
#pragma unroll
    for (int i = 0; i < 2; ++i) aF[i] = frag_ld(As + (wm * 32 + i * 16) * 40, 40);
#pragma unroll
    for (int j = 0; j < 4; ++j) bF[j] = frag_ld(Bs + (wn * 64 + j * 16) * 40, 40);
#pragma unroll
    for (int i = 0; i < 2; ++i)
#pragma unroll
      for (int j = 0; j < 4; ++j) acc[i][j] = wmma32(aF[i], bF[j], acc[i][j]);
    __syncthreads();
  }

  // Epilogue. C/D layout: VGPR r holds rows (r, r+8) split per lane half.
  const int half = lane >> 4, ln = lane & 15;
#pragma unroll
  for (int i = 0; i < 2; ++i)
#pragma unroll
    for (int j = 0; j < 4; ++j) {
      const int mb = m0 + wm * 32 + i * 16 + half * 8;
      const int n  = n0 + wn * 64 + j * 16 + ln;
#pragma unroll
      for (int r = 0; r < 8; ++r) {
        const int m = mb + r;
        __bf16 v = cvt_bf16(acc[i][j][r]);
        if (z == 0) {
          Qo[(size_t)m * D_ + n] = v;
        } else if (z == 1) {
          Ko[(size_t)m * D_ + n] = v;
        } else {
          // Raw-reshape head view: flat [B*H*S x 128]; group = rh >> 11.
          size_t flat = (size_t)m * D_ + n;
          int rh = (int)(flat >> 7);
          int c  = (int)(flat & 127);
          int g  = rh >> 11;
          int s  = rh & (S_ - 1);
          Vt[(((size_t)g * CH_ + c) << 11) + s] = v;   // Vt[g][c][s]
        }
      }
    }
}

// ---------------------------------------------------------------------------
// Kernel 2: flash attention per (group, 128-query tile).
// Q/K viewed as flat [B*H*S x 128] bf16; groups = 2048-row blocks.
// Tiles staged with GLOBAL_LOAD_ASYNC_TO_LDS_B128 when available.
// Per wave: 16 q-rows x full 128-key block; online softmax with half-wave
// shuffles; P re-staged to LDS bf16 and fed back as WMMA A operand.
// ---------------------------------------------------------------------------
__global__ __launch_bounds__(256) void attn_kernel(
    const __bf16* __restrict__ Qh, const __bf16* __restrict__ Kh,
    const __bf16* __restrict__ Vt, __bf16* __restrict__ ctx)
{
  extern __shared__ __bf16 smem[];
  __bf16* Qs = smem;                 // 128 x 136
  __bf16* Ks = Qs + 128 * 136;       // 128 keys x 128 dims
  __bf16* Vs = Ks + 128 * 136;       // V^T: 128 dims x 128 keys
  __bf16* Ps = Vs + 128 * 136;       // 128 q x 128 keys (wave-private rows)

  const int t = (int)threadIdx.x;
  const int lane = t & 31, wave = t >> 5;
  const int half = lane >> 4, ln = lane & 15;
  const int g = (int)blockIdx.y;
  const int q0 = (int)blockIdx.x * 128;
  const size_t grow = (size_t)g * S_;
  const int prow = wave * 16;

  // stage Q tile once (async; completion covered by first in-loop wait)
#pragma unroll
  for (int i = 0; i < 8; ++i) {
    int e = t + i * 256;               // 2048 chunks = 128 rows x 16
    int r = e >> 4, c8 = e & 15;
    cp16_g2l(Qs + r * 136 + c8 * 8,
             Qh + (grow + q0 + r) * CH_ + c8 * 8);
  }

  float rowmax[8], rowsum[8];
  v8f cacc[8];
#pragma unroll
  for (int j = 0; j < 8; ++j) { rowmax[j] = -3.0e38f; rowsum[j] = 0.f; }
#pragma unroll
  for (int nt = 0; nt < 8; ++nt) cacc[nt] = v8f_zero();

  for (int kb = 0; kb < S_; kb += 128) {
#pragma unroll
    for (int i = 0; i < 8; ++i) {      // stage K tile + V^T tile
      int e = t + i * 256;
      int r = e >> 4, c8 = e & 15;
      cp16_g2l(Ks + r * 136 + c8 * 8,
               Kh + (grow + kb + r) * CH_ + c8 * 8);
      cp16_g2l(Vs + r * 136 + c8 * 8,
               Vt + ((size_t)g * CH_ + r) * S_ + kb + c8 * 8);
    }
    cp_g2l_wait();                     // s_wait_asynccnt 0
    __syncthreads();

    // scores: S = Q K^T  (both operands contiguous rows -> same frag_ld)
    v8f sacc[8];
#pragma unroll
    for (int nt = 0; nt < 8; ++nt) sacc[nt] = v8f_zero();
#pragma unroll
    for (int kk = 0; kk < 4; ++kk) {
      v16bf aQ = frag_ld(Qs + prow * 136 + kk * 32, 136);
#pragma unroll
      for (int nt = 0; nt < 8; ++nt) {
        v16bf bK = frag_ld(Ks + nt * 16 * 136 + kk * 32, 136);
        sacc[nt] = wmma32(aQ, bK, sacc[nt]);
      }
    }
#pragma unroll
    for (int nt = 0; nt < 8; ++nt) sacc[nt] *= SCALE_;

    // online softmax; VGPR j tracks row j (lanes 0-15) / j+8 (lanes 16-31)
#pragma unroll
    for (int j = 0; j < 8; ++j) {
      float tm = sacc[0][j];
#pragma unroll
      for (int nt = 1; nt < 8; ++nt) tm = fmaxf(tm, sacc[nt][j]);
#pragma unroll
      for (int off = 1; off < 16; off <<= 1)
        tm = fmaxf(tm, __shfl_xor(tm, off, 32));
      float mn = fmaxf(rowmax[j], tm);
      float alpha = __expf(rowmax[j] - mn);
      rowmax[j] = mn;
      float ts = 0.f;
#pragma unroll
      for (int nt = 0; nt < 8; ++nt) {
        float p = __expf(sacc[nt][j] - mn);
        ts += p;
        Ps[(prow + j + half * 8) * 136 + nt * 16 + ln] = cvt_bf16(p);
      }
#pragma unroll
      for (int off = 1; off < 16; off <<= 1)
        ts += __shfl_xor(ts, off, 32);
      rowsum[j] = rowsum[j] * alpha + ts;
#pragma unroll
      for (int nt = 0; nt < 8; ++nt) cacc[nt][j] *= alpha;
    }
    __syncthreads();   // order P stores before P fragment loads

    // ctx += P @ V   (B operand = V^T rows = dims, keys contiguous)
#pragma unroll
    for (int kk = 0; kk < 4; ++kk) {
      v16bf aP = frag_ld(Ps + prow * 136 + kk * 32, 136);
#pragma unroll
      for (int nt = 0; nt < 8; ++nt) {
        v16bf bV = frag_ld(Vs + nt * 16 * 136 + kk * 32, 136);
        cacc[nt] = wmma32(aP, bV, cacc[nt]);
      }
    }
    __syncthreads();   // before next iteration restages Ks/Vs
  }

  // normalize + write ctx (flat [B*H*S x 128] bf16 == [B*S x D] view)
#pragma unroll
  for (int j = 0; j < 8; ++j) {
    float inv = 1.0f / rowsum[j];
    size_t row = grow + q0 + prow + j + half * 8;
#pragma unroll
    for (int nt = 0; nt < 8; ++nt)
      ctx[row * CH_ + nt * 16 + ln] = cvt_bf16(cacc[nt][j] * inv);
  }
}

// ---------------------------------------------------------------------------
// Kernel 3: out = ctx @ Wp.T + bp + residual(query), fp32 to d_out.
// ---------------------------------------------------------------------------
__global__ __launch_bounds__(256) void proj_kernel(
    const __bf16* __restrict__ ctx, const float* __restrict__ Wp,
    const float* __restrict__ bp, const float* __restrict__ resid,
    float* __restrict__ out)
{
  __shared__ __bf16 As[128 * 40];
  __shared__ __bf16 Bs[128 * 40];

  const int t = (int)threadIdx.x;
  const int lane = t & 31, wave = t >> 5;
  const int wm = wave & 3, wn = wave >> 2;
  const int m0 = (int)blockIdx.x * 128;
  const int n0 = (int)blockIdx.y * 128;

  v8f acc[2][4];
#pragma unroll
  for (int i = 0; i < 2; ++i)
#pragma unroll
    for (int j = 0; j < 4; ++j) acc[i][j] = v8f_zero();

  for (int k0 = 0; k0 < D_; k0 += 32) {
#pragma unroll
    for (int i = 0; i < 2; ++i) {        // A already bf16: async copy
      int e = t + i * 256;               // 512 chunks = 128 rows x 4
      int r = e >> 2, c8 = e & 3;
      cp16_g2l(As + r * 40 + c8 * 8,
               ctx + (size_t)(m0 + r) * D_ + k0 + c8 * 8);
    }
    // Wp fp32 -> bf16: batch loads, then convert.
    f32x4 wv[4];
#pragma unroll
    for (int i = 0; i < 4; ++i) {
      int e = t + i * 256;
      int r = e >> 3, c4 = e & 7;
      wv[i] = *(const f32x4*)(Wp + (size_t)(n0 + r) * D_ + k0 + c4 * 4);
    }
#pragma unroll
    for (int i = 0; i < 4; ++i) {
      int e = t + i * 256;
      int r = e >> 3, c4 = e & 7;
      *(bf16x4*)(Bs + r * 40 + c4 * 4) = cvt_bf16x4(wv[i]);
    }
    if (k0 + 32 < D_)
      __builtin_prefetch(Wp + (size_t)(n0 + (t >> 1)) * D_ + k0 + 32, 0, 1);
    cp_g2l_wait();
    __syncthreads();

    v16bf aF[2], bF[4];
#pragma unroll
    for (int i = 0; i < 2; ++i) aF[i] = frag_ld(As + (wm * 32 + i * 16) * 40, 40);
#pragma unroll
    for (int j = 0; j < 4; ++j) bF[j] = frag_ld(Bs + (wn * 64 + j * 16) * 40, 40);
#pragma unroll
    for (int i = 0; i < 2; ++i)
#pragma unroll
      for (int j = 0; j < 4; ++j) acc[i][j] = wmma32(aF[i], bF[j], acc[i][j]);
    __syncthreads();
  }

  const int half = lane >> 4, ln = lane & 15;
#pragma unroll
  for (int i = 0; i < 2; ++i)
#pragma unroll
    for (int j = 0; j < 4; ++j) {
      const int mb = m0 + wm * 32 + i * 16 + half * 8;
      const int n  = n0 + wn * 64 + j * 16 + ln;
      const float bias = bp[n];
#pragma unroll
      for (int r = 0; r < 8; ++r) {
        const int m = mb + r;
        out[(size_t)m * D_ + n] =
            acc[i][j][r] + bias + resid[(size_t)m * D_ + n];
      }
    }
}

// ---------------------------------------------------------------------------
// Kernel 4: in-place LayerNorm over D=1024 (one block per row).
// ---------------------------------------------------------------------------
__global__ __launch_bounds__(256) void ln_kernel(
    float* __restrict__ out, const float* __restrict__ gamma,
    const float* __restrict__ beta)
{
  __shared__ float rs[8], rq[8];
  const int t = (int)threadIdx.x, lane = t & 31, wave = t >> 5;
  float* p = out + (size_t)blockIdx.x * D_;

  f32x4 v = *(const f32x4*)(p + t * 4);
  float s = v.x + v.y + v.z + v.w;
  float q = v.x * v.x + v.y * v.y + v.z * v.z + v.w * v.w;
#pragma unroll
  for (int off = 1; off < 32; off <<= 1) {
    s += __shfl_xor(s, off, 32);
    q += __shfl_xor(q, off, 32);
  }
  if (lane == 0) { rs[wave] = s; rq[wave] = q; }
  __syncthreads();
  float S = 0.f, Q = 0.f;
#pragma unroll
  for (int w = 0; w < 8; ++w) { S += rs[w]; Q += rq[w]; }
  const float mu  = S * (1.0f / D_);
  const float var = Q * (1.0f / D_) - mu * mu;
  const float rr  = rsqrtf(var + LN_EPS_);

  f32x4 gm = *(const f32x4*)(gamma + t * 4);
  f32x4 bt = *(const f32x4*)(beta + t * 4);
  f32x4 o;
  o.x = (v.x - mu) * rr * gm.x + bt.x;
  o.y = (v.y - mu) * rr * gm.y + bt.y;
  o.z = (v.z - mu) * rr * gm.z + bt.z;
  o.w = (v.w - mu) * rr * gm.w + bt.w;
  *(f32x4*)(p + t * 4) = o;
}

// ---------------------------------------------------------------------------
extern "C" void kernel_launch(void* const* d_in, const int* in_sizes, int n_in,
                              void* d_out, int out_size, void* d_ws, size_t ws_size,
                              hipStream_t stream) {
  const float* query = (const float*)d_in[0];
  const float* keys  = (const float*)d_in[1];
  const float* Wq    = (const float*)d_in[2];
  const float* Wk    = (const float*)d_in[3];
  const float* Wv    = (const float*)d_in[4];
  const float* Wp    = (const float*)d_in[5];
  const float* bp    = (const float*)d_in[6];
  const float* gamma = (const float*)d_in[7];
  const float* beta  = (const float*)d_in[8];
  float* out = (float*)d_out;

  const size_t elems = (size_t)MROWS_ * D_;          // 16,777,216
  char* ws = (char*)d_ws;                            // 4 x 32MiB bf16 buffers
  __bf16* Qb  = (__bf16*)(ws + 0 * elems * 2);
  __bf16* Kb  = (__bf16*)(ws + 1 * elems * 2);
  __bf16* Vtb = (__bf16*)(ws + 2 * elems * 2);
  __bf16* Cb  = (__bf16*)(ws + 3 * elems * 2);

  dim3 gq(MROWS_ / 128, D_ / 128, 3);
  qkv_kernel<<<gq, 256, 0, stream>>>(query, keys, Wq, Wk, Wv, Qb, Kb, Vtb);

  dim3 ga(S_ / 128, GROUPS_);
  const size_t attn_lds = (size_t)4 * 128 * 136 * sizeof(__bf16);  // 139,264 B
  attn_kernel<<<ga, 256, attn_lds, stream>>>(Qb, Kb, Vtb, Cb);

  dim3 gp(MROWS_ / 128, D_ / 128);
  proj_kernel<<<gp, 256, 0, stream>>>(Cb, Wp, bp, query, out);

  ln_kernel<<<MROWS_, 256, 0, stream>>>(out, gamma, beta);

  (void)in_sizes; (void)n_in; (void)out_size; (void)ws_size;
}
